// Weight_82884278878349
// MI455X (gfx1250) — matrix operand order repaired and checked by reading, not compiled, not run on hardware
//
#include <hip/hip_runtime.h>
#include <hip/hip_bf16.h>
#include <stdint.h>

// ---------------------------------------------------------------------------
// Types for WMMA
// ---------------------------------------------------------------------------
typedef __attribute__((ext_vector_type(16))) _Float16 v16h;
typedef __attribute__((ext_vector_type(8)))  _Float16 v8h;
typedef __attribute__((ext_vector_type(8)))  float    v8f;

#define BB 16
#define VD 2048
#define VN 49
#define VNP 64          // padded token dim for vision (49 -> 64)
#define AN 64
#define AD 128
#define VOUT_ELEMS (BB*VN*VD)   // 1,605,632
#define AOUT_ELEMS (BB*AN*AD)   // 131,072

__device__ __forceinline__ v8f wmma_f16(v16h a, v16h b, v8f c) {
  // D = A(16x32 f16) * B(32x16 f16) + C(16x16 f32)
  return __builtin_amdgcn_wmma_f32_16x16x32_f16(
      /*neg_a=*/false, a, /*neg_b=*/false, b,
      /*c_mod=*/(short)0, c, /*reuse_a=*/false, /*reuse_b=*/false);
}

// A operand: 16x32 f16 tile, row-major source with leading dim `ld` (halves).
// Layout: lanes 0-15 row M=lane, halves 0..15=K0..15;
//         lanes 16-31 row M=lane-16, halves 0..15=K16..31.
__device__ __forceinline__ v16h load_A16x32(const _Float16* tile, int ld) {
  int lane = threadIdx.x & 31;
  int r = lane & 15;
  int k = (lane & 16) ? 16 : 0;
  const _Float16* p = tile + (size_t)r * ld + k;
  union { v16h v; v8h h[2]; } u;
  u.h[0] = *(const v8h*)(p);
  u.h[1] = *(const v8h*)(p + 8);
  return u.v;
}

// B operand: 32x16 f16 tile, source stored [col][K] (col-major over N) with
// leading dim `ld` halves. Layout (contiguous k, same as A): lanes 0-15 col N=lane, halves=K0..15;
// lanes 16-31 col N=lane-16, halves=K16..31.
__device__ __forceinline__ v16h load_B32x16(const _Float16* tileT, int ld) {
  int lane = threadIdx.x & 31;
  int c = lane & 15;
  int k = (lane & 16) ? 16 : 0;
  return *(const v16h*)(tileT + (size_t)c * ld + k);
}

// Async global->LDS copy of 16 bytes per lane (CDNA5 ASYNCcnt path).
__device__ __forceinline__ void async_copy_b128(uint32_t lds_addr, uint64_t gaddr) {
  asm volatile("global_load_async_to_lds_b128 %0, %1, off"
               :: "v"(lds_addr), "v"(gaddr) : "memory");
}
__device__ __forceinline__ void wait_asynccnt0() {
  asm volatile("s_wait_asynccnt 0x0" ::: "memory");
}

// ---------------------------------------------------------------------------
// Generic BN (training-mode batch stats over axes (0,2) of a (B,C,L) tensor)
// ---------------------------------------------------------------------------
__global__ void bn_stats_kernel(const float* __restrict__ x, float* mean,
                                float* rstd, int Bn, int C, int L) {
  int c = blockIdx.x;
  int N = Bn * L;
  float s = 0.f, s2 = 0.f;
  for (int i = threadIdx.x; i < N; i += blockDim.x) {
    int b = i / L, l = i - b * L;
    float v = x[(size_t)b * C * L + (size_t)c * L + l];
    s += v; s2 += v * v;
  }
  __shared__ float sh[256], sh2[256];
  sh[threadIdx.x] = s; sh2[threadIdx.x] = s2;
  __syncthreads();
  for (int o = blockDim.x >> 1; o > 0; o >>= 1) {
    if ((int)threadIdx.x < o) {
      sh[threadIdx.x] += sh[threadIdx.x + o];
      sh2[threadIdx.x] += sh2[threadIdx.x + o];
    }
    __syncthreads();
  }
  if (threadIdx.x == 0) {
    float m = sh[0] / (float)N;
    float var = sh2[0] / (float)N - m * m;
    mean[c] = m;
    rstd[c] = rsqrtf(var + 1e-5f);
  }
}

__global__ void bn_apply_relu_kernel(float* x, const float* __restrict__ mean,
                                     const float* __restrict__ rstd,
                                     const float* __restrict__ g,
                                     const float* __restrict__ bta,
                                     int total, int C, int L) {
  int i = blockIdx.x * blockDim.x + threadIdx.x;
  if (i >= total) return;
  int c = (i / L) % C;
  float v = (x[i] - mean[c]) * rstd[c] * g[c] + bta[c];
  x[i] = v > 0.f ? v : 0.f;
}

// ---------------------------------------------------------------------------
// Vision front path (small GEMMs, memory bound)
// ---------------------------------------------------------------------------
// vh_pre[b,n,h] = sum_d Vision[b,d,n] * vW1[h,d] + vb1[h]     (B,49,49)
__global__ void vh_pre_kernel(const float* __restrict__ Vis,
                              const float* __restrict__ W,
                              const float* __restrict__ bias, float* out) {
  int i = blockIdx.x * blockDim.x + threadIdx.x;
  if (i >= BB * VN * VN) return;
  int h = i % VN, n = (i / VN) % VN, b = i / (VN * VN);
  const float* vp = Vis + (size_t)b * VD * VN + n;      // stride VN over d
  const float* wp = W + (size_t)h * VD;
  float acc = bias[h];
  for (int d = 0; d < VD; ++d) acc += vp[(size_t)d * VN] * wp[d];
  out[i] = acc;
}

// vquery[b,n,e] = sum_h vh[b,n,h] * vW2[e,h] + vb2[e]         (B,49,2048)
__global__ void vquery_kernel(const float* __restrict__ vh,
                              const float* __restrict__ W,
                              const float* __restrict__ bias, float* out) {
  int i = blockIdx.x * blockDim.x + threadIdx.x;
  if (i >= BB * VN * VD) return;
  int e = i % VD, n = (i / VD) % VN, b = i / (VN * VD);
  const float* hp = vh + ((size_t)b * VN + n) * VN;
  const float* wp = W + (size_t)e * VN;
  float acc = bias[e];
  for (int h = 0; h < VN; ++h) acc += hp[h] * wp[h];
  out[i] = acc;
}

// vkeys_pre[b,o,d] = sum_n kvW[o,n] * Vision[b,d,n] + kvb[o]  (B,49,2048)
__global__ void vkeys_pre_kernel(const float* __restrict__ Vis,
                                 const float* __restrict__ W,
                                 const float* __restrict__ bias, float* out) {
  int i = blockIdx.x * blockDim.x + threadIdx.x;
  if (i >= BB * VN * VD) return;
  int d = i % VD, o = (i / VD) % VN, b = i / (VN * VD);
  const float* vp = Vis + (size_t)b * VD * VN + (size_t)d * VN;  // contiguous n
  const float* wp = W + (size_t)o * VN;
  float acc = bias[o];
  for (int n = 0; n < VN; ++n) acc += wp[n] * vp[n];
  out[i] = acc;
}

// Pack to f16, n padded to 64 with zeros:
//   Qt[b][d][n] = vkeys[b,n,d]   (2048 x 64)
//   Kt[b][e][n] = vquery[b,n,e]  (2048 x 64)
//   Vp[b][n][d] = Vision[b,d,n]  (64 x 2048)
__global__ void pack_kernel(const float* __restrict__ vkeys,
                            const float* __restrict__ vquery,
                            const float* __restrict__ Vis,
                            _Float16* Qt, _Float16* Kt, _Float16* Vp) {
  int i = blockIdx.x * blockDim.x + threadIdx.x;
  if (i >= BB * VD * VNP) return;
  int j = i % VNP, r = (i / VNP) % VD, b = i / (VNP * VD);
  float q = 0.f, k = 0.f, v = 0.f;
  if (j < VN) {
    size_t idx = ((size_t)b * VN + j) * VD + r;
    q = vkeys[idx];
    k = vquery[idx];
    v = Vis[(size_t)b * VD * VN + (size_t)r * VN + j];
  }
  Qt[((size_t)b * VD + r) * VNP + j] = (_Float16)q;
  Kt[((size_t)b * VD + r) * VNP + j] = (_Float16)k;
  Vp[((size_t)b * VNP + j) * VD + r] = (_Float16)v;
}

// ---------------------------------------------------------------------------
// Pass 1: per-row softmax stats (max, sum-exp) for scores_v.
// 1 wave per (b, 16-row d-tile). S = Qt(16x64) x Kt^T(64x16) via 2 WMMAs/tile.
// ---------------------------------------------------------------------------
__global__ __launch_bounds__(32)
void rowstats_kernel(const _Float16* __restrict__ Qt,
                     const _Float16* __restrict__ Kt,
                     float* __restrict__ rowm, float* __restrict__ rowz) {
  const float scale = 0.022097086912079608f;  // 1/sqrt(2048)
  int b = blockIdx.x >> 7;
  int d0 = (blockIdx.x & 127) << 4;
  int lane = threadIdx.x & 31;

  const _Float16* qb = Qt + ((size_t)b * VD + d0) * VNP;
  v16h a0 = load_A16x32(qb, VNP);
  v16h a1 = load_A16x32(qb + 32, VNP);

  float rmax[8], rsum[8];
#pragma unroll
  for (int i = 0; i < 8; ++i) { rmax[i] = -3.0e38f; rsum[i] = 0.f; }

  for (int e0 = 0; e0 < VD; e0 += 16) {
    const _Float16* kb = Kt + ((size_t)b * VD + e0) * VNP;
    v16h b0 = load_B32x16(kb, VNP);
    v16h b1 = load_B32x16(kb + 32, VNP);
    v8f c = {};
    c = wmma_f16(a0, b0, c);
    c = wmma_f16(a1, b1, c);
#pragma unroll
    for (int i = 0; i < 8; ++i) {
      float x = c[i] * scale;
      float nm = fmaxf(rmax[i], x);
      rsum[i] = rsum[i] * __expf(rmax[i] - nm) + __expf(x - nm);
      rmax[i] = nm;
    }
  }
  // combine the 16 e-columns held across lanes of each half-wave
  for (int mask = 1; mask < 16; mask <<= 1) {
#pragma unroll
    for (int i = 0; i < 8; ++i) {
      float om = __shfl_xor(rmax[i], mask, 32);
      float os = __shfl_xor(rsum[i], mask, 32);
      float nm = fmaxf(rmax[i], om);
      rsum[i] = rsum[i] * __expf(rmax[i] - nm) + os * __expf(om - nm);
      rmax[i] = nm;
    }
  }
  if ((lane & 15) == 0) {
    int rbase = d0 + ((lane & 16) ? 8 : 0);
#pragma unroll
    for (int i = 0; i < 8; ++i) {
      rowm[(size_t)b * VD + rbase + i] = rmax[i];
      rowz[(size_t)b * VD + rbase + i] = rsum[i];
    }
  }
}

// ---------------------------------------------------------------------------
// Pass 2: fused P recompute + v_out accumulation.
// Block = 4 waves, covers one (b, 64-wide e-block). Loops d in steps of 32:
//   - the 32x64 Qt d-window (4KB, contiguous) is staged into LDS with
//     double-buffered async global->LDS copies (ASYNCcnt), overlapping the
//     copy of window i+1 with compute of window i.
//   phase A: wave w computes S tiles for e-tile w (2 d-subtiles) from LDS Qs,
//            forms P = exp(s*scale - m_d)/Z_d, stores f16 transposed to LDS.
//   phase B: wave w owns n-tile w, does out += Vp(16x32) x P(32x16) per e-tile.
// ---------------------------------------------------------------------------
__global__ __launch_bounds__(128)
void vattn_kernel(const _Float16* __restrict__ Qt,
                  const _Float16* __restrict__ Kt,
                  const _Float16* __restrict__ Vp,
                  const float* __restrict__ rowm,
                  const float* __restrict__ rowz,
                  float* __restrict__ vout) {
  const float scale = 0.022097086912079608f;  // 1/sqrt(2048)
  int b = blockIdx.x >> 5;
  int e0 = (blockIdx.x & 31) << 6;
  int wave = threadIdx.x >> 5;
  int lane = threadIdx.x & 31;

  __shared__ __align__(32) _Float16 Qs[2][32][VNP];   // double-buffered Qt window
  __shared__ __align__(32) _Float16 PhT[64][32];      // [e_local][d_local]
  __shared__ float sm[32], sz[32];

  const _Float16* Qb = Qt + (size_t)b * VD * VNP;
  const _Float16* Kb = Kt + (size_t)b * VD * VNP;
  const _Float16* Vb = Vp + (size_t)b * VNP * VD;

  // B operand of the score GEMM is constant over the d loop (this wave's e-tile)
  const _Float16* kb = Kb + (size_t)(e0 + wave * 16) * VNP;
  v16h kb0 = load_B32x16(kb, VNP);
  v16h kb1 = load_B32x16(kb + 32, VNP);

  v8f acc0 = {}, acc1 = {}, acc2 = {}, acc3 = {};

  // issue async copy of the first Qt window: 4KB = 256 x 16B chunks,
  // 128 threads x 2 chunks each
  {
    uint64_t g = (uint64_t)(uintptr_t)Qb;  // window d0=0 starts at Qb
    uint32_t l = (uint32_t)(uintptr_t)&Qs[0][0][0];
    async_copy_b128(l + threadIdx.x * 16u,          g + threadIdx.x * 16u);
    async_copy_b128(l + (threadIdx.x + 128u) * 16u, g + (threadIdx.x + 128u) * 16u);
  }

  int buf = 0;
  for (int d0 = 0; d0 < VD; d0 += 32) {
    if (threadIdx.x < 32) {
      sm[threadIdx.x] = rowm[(size_t)b * VD + d0 + threadIdx.x];
      sz[threadIdx.x] = 1.0f / rowz[(size_t)b * VD + d0 + threadIdx.x];
    }
    wait_asynccnt0();        // Qs[buf] landed (this wave's async loads)
    __syncthreads();         // ... and everyone else's

    // kick off the copy of the next window into the other buffer
    if (d0 + 32 < VD) {
      uint64_t g = (uint64_t)(uintptr_t)(Qb + (size_t)(d0 + 32) * VNP);
      uint32_t l = (uint32_t)(uintptr_t)&Qs[buf ^ 1][0][0];
      async_copy_b128(l + threadIdx.x * 16u,          g + threadIdx.x * 16u);
      async_copy_b128(l + (threadIdx.x + 128u) * 16u, g + (threadIdx.x + 128u) * 16u);
    }

    // ---- phase A: scores -> P tile (32 x 64) in LDS, f16 transposed ----
#pragma unroll
    for (int dt = 0; dt < 2; ++dt) {
      const _Float16* qtile = &Qs[buf][dt * 16][0];
      v16h a0 = load_A16x32(qtile, VNP);
      v16h a1 = load_A16x32(qtile + 32, VNP);
      v8f c = {};
      c = wmma_f16(a0, kb0, c);
      c = wmma_f16(a1, kb1, c);
      int rbase = dt * 16 + ((lane & 16) ? 8 : 0);
      v8h ph;
#pragma unroll
      for (int i = 0; i < 8; ++i) {
        int r = rbase + i;
        float p = __expf(c[i] * scale - sm[r]) * sz[r];
        ph[i] = (_Float16)p;
      }
      *(v8h*)&PhT[wave * 16 + (lane & 15)][rbase] = ph;
    }
    __syncthreads();

    // ---- phase B: out(n-tile=wave, 4 e-tiles) += Vp x P ----
    const _Float16* vtile = Vb + (size_t)(wave * 16) * VD + d0;
    if (d0 + 32 < VD) __builtin_prefetch(vtile + 32, 0, 0);
    v16h av = load_A16x32(vtile, VD);
    acc0 = wmma_f16(av, load_B32x16(&PhT[0][0], 32), acc0);
    acc1 = wmma_f16(av, load_B32x16(&PhT[16][0], 32), acc1);
    acc2 = wmma_f16(av, load_B32x16(&PhT[32][0], 32), acc2);
    acc3 = wmma_f16(av, load_B32x16(&PhT[48][0], 32), acc3);
    __syncthreads();
    buf ^= 1;
  }

  // store v_out (skip padded rows n >= 49)
  int ecol = lane & 15;
  int nbase = wave * 16 + ((lane & 16) ? 8 : 0);
#pragma unroll
  for (int i = 0; i < 8; ++i) {
    int n = nbase + i;
    if (n < VN) {
      float* o = vout + ((size_t)b * VN + n) * VD + e0 + ecol;
      o[0]  = acc0[i];
      o[16] = acc1[i];
      o[32] = acc2[i];
      o[48] = acc3[i];
    }
  }
}

// ---------------------------------------------------------------------------
// Audio path (tiny; plain VALU kernels)
// ---------------------------------------------------------------------------
__global__ void ah_pre_kernel(const float* __restrict__ A,
                              const float* __restrict__ W,
                              const float* __restrict__ bias, float* out) {
  int i = blockIdx.x * blockDim.x + threadIdx.x;
  if (i >= BB * AN * AN) return;
  int h = i % AN, n = (i / AN) % AN, b = i / (AN * AN);
  const float* ap = A + ((size_t)b * AN + n) * AD;
  const float* wp = W + (size_t)h * AD;
  float acc = bias[h];
  for (int d = 0; d < AD; ++d) acc += ap[d] * wp[d];
  out[i] = acc;
}

__global__ void aquery_kernel(const float* __restrict__ ah,
                              const float* __restrict__ W,
                              const float* __restrict__ bias, float* out) {
  int i = blockIdx.x * blockDim.x + threadIdx.x;
  if (i >= BB * AN * AD) return;
  int e = i % AD, n = (i / AD) % AN, b = i / (AN * AD);
  const float* hp = ah + ((size_t)b * AN + n) * AN;
  const float* wp = W + (size_t)e * AN;
  float acc = bias[e];
  for (int h = 0; h < AN; ++h) acc += hp[h] * wp[h];
  out[i] = acc;
}

__global__ void akeys_pre_kernel(const float* __restrict__ A,
                                 const float* __restrict__ W,
                                 const float* __restrict__ bias, float* out) {
  int i = blockIdx.x * blockDim.x + threadIdx.x;
  if (i >= BB * AN * AD) return;
  int d = i % AD, o = (i / AD) % AN, b = i / (AN * AD);
  const float* wp = W + (size_t)o * AN;
  const float* ap = A + (size_t)b * AN * AD + d;   // stride AD over n
  float acc = bias[o];
  for (int n = 0; n < AN; ++n) acc += wp[n] * ap[(size_t)n * AD];
  out[i] = acc;
}

__global__ void scores_a_kernel(const float* __restrict__ akeys,
                                const float* __restrict__ aquery,
                                float* out) {
  int i = blockIdx.x * blockDim.x + threadIdx.x;
  if (i >= BB * AD * AD) return;
  int e = i % AD, d = (i / AD) % AD, b = i / (AD * AD);
  const float* kp = akeys + (size_t)b * AN * AD + d;
  const float* qp = aquery + (size_t)b * AN * AD + e;
  float acc = 0.f;
  for (int n = 0; n < AN; ++n)
    acc += kp[(size_t)n * AD] * qp[(size_t)n * AD];
  out[i] = acc * 0.08838834764831845f;  // 1/sqrt(128)
}

__global__ __launch_bounds__(128)
void softmax128_kernel(float* s) {
  float* p = s + (size_t)blockIdx.x * 128;
  float v = p[threadIdx.x];
  __shared__ float sh[128];
  sh[threadIdx.x] = v;
  __syncthreads();
  for (int o = 64; o > 0; o >>= 1) {
    if ((int)threadIdx.x < o) sh[threadIdx.x] = fmaxf(sh[threadIdx.x], sh[threadIdx.x + o]);
    __syncthreads();
  }
  float m = sh[0];
  __syncthreads();
  float e = __expf(v - m);
  sh[threadIdx.x] = e;
  __syncthreads();
  for (int o = 64; o > 0; o >>= 1) {
    if ((int)threadIdx.x < o) sh[threadIdx.x] += sh[threadIdx.x + o];
    __syncthreads();
  }
  p[threadIdx.x] = e / sh[0];
}

__global__ void aout_kernel(const float* __restrict__ A,
                            const float* __restrict__ sc, float* out) {
  int i = blockIdx.x * blockDim.x + threadIdx.x;
  if (i >= BB * AN * AD) return;
  int e = i % AD, n = (i / AD) % AN, b = i / (AN * AD);
  const float* ap = A + ((size_t)b * AN + n) * AD;
  const float* sp = sc + (size_t)b * AD * AD + e;   // stride AD over d
  float acc = 0.f;
  for (int d = 0; d < AD; ++d) acc += ap[d] * sp[(size_t)d * AD];
  out[i] = acc;
}

// ---------------------------------------------------------------------------
// Launch
// ---------------------------------------------------------------------------
extern "C" void kernel_launch(void* const* d_in, const int* in_sizes, int n_in,
                              void* d_out, int out_size, void* d_ws, size_t ws_size,
                              hipStream_t stream) {
  (void)in_sizes; (void)n_in; (void)out_size; (void)ws_size;
  const float* Vis  = (const float*)d_in[0];
  const float* Aud  = (const float*)d_in[1];
  const float* aW1  = (const float*)d_in[2];
  const float* ab1  = (const float*)d_in[3];
  const float* a_g1 = (const float*)d_in[4];
  const float* a_b1 = (const float*)d_in[5];
  const float* aW2  = (const float*)d_in[6];
  const float* ab2  = (const float*)d_in[7];
  const float* vW1  = (const float*)d_in[8];
  const float* vb1  = (const float*)d_in[9];
  const float* v_g1 = (const float*)d_in[10];
  const float* v_b1 = (const float*)d_in[11];
  const float* vW2  = (const float*)d_in[12];
  const float* vb2  = (const float*)d_in[13];
  const float* kaW  = (const float*)d_in[14];
  const float* kab  = (const float*)d_in[15];
  const float* ka_g = (const float*)d_in[16];
  const float* ka_b = (const float*)d_in[17];
  const float* kvW  = (const float*)d_in[18];
  const float* kvb  = (const float*)d_in[19];
  const float* kv_g = (const float*)d_in[20];
  const float* kv_b = (const float*)d_in[21];
  // d_in[22]=embed_a, d_in[23]=embed_v are unused by the output

  float* vout = (float*)d_out;
  float* aout = (float*)d_out + VOUT_ELEMS;

  char* w = (char*)d_ws;
  size_t off = 0;
  auto alloc = [&](size_t bytes) -> void* {
    void* p = w + off;
    off = (off + bytes + 255) & ~(size_t)255;
    return p;
  };
  float*    vh     = (float*)alloc((size_t)BB * VN * VN * 4);
  float*    vquery = (float*)alloc((size_t)BB * VN * VD * 4);
  float*    vkeys  = (float*)alloc((size_t)BB * VN * VD * 4);
  _Float16* Qt     = (_Float16*)alloc((size_t)BB * VD * VNP * 2);
  _Float16* Kt     = (_Float16*)alloc((size_t)BB * VD * VNP * 2);
  _Float16* Vp     = (_Float16*)alloc((size_t)BB * VNP * VD * 2);
  float*    rowm   = (float*)alloc((size_t)BB * VD * 4);
  float*    rowz   = (float*)alloc((size_t)BB * VD * 4);
  float*    ah     = (float*)alloc((size_t)BB * AN * AN * 4);
  float*    aquery = (float*)alloc((size_t)BB * AN * AD * 4);
  float*    akeys  = (float*)alloc((size_t)BB * AN * AD * 4);
  float*    sca    = (float*)alloc((size_t)BB * AD * AD * 4);
  float*    st_v1m = (float*)alloc(64 * 4);
  float*    st_v1r = (float*)alloc(64 * 4);
  float*    st_kvm = (float*)alloc(64 * 4);
  float*    st_kvr = (float*)alloc(64 * 4);
  float*    st_a1m = (float*)alloc(64 * 4);
  float*    st_a1r = (float*)alloc(64 * 4);
  float*    st_kam = (float*)alloc(64 * 4);
  float*    st_kar = (float*)alloc(64 * 4);

  const int T = 256;
  // ---------- vision path ----------
  vh_pre_kernel<<<(BB*VN*VN + T - 1) / T, T, 0, stream>>>(Vis, vW1, vb1, vh);
  bn_stats_kernel<<<VN, T, 0, stream>>>(vh, st_v1m, st_v1r, BB, VN, VN);
  bn_apply_relu_kernel<<<(BB*VN*VN + T - 1) / T, T, 0, stream>>>(
      vh, st_v1m, st_v1r, v_g1, v_b1, BB*VN*VN, VN, VN);
  vquery_kernel<<<(BB*VN*VD + T - 1) / T, T, 0, stream>>>(vh, vW2, vb2, vquery);
  vkeys_pre_kernel<<<(BB*VN*VD + T - 1) / T, T, 0, stream>>>(Vis, kvW, kvb, vkeys);
  bn_stats_kernel<<<VN, T, 0, stream>>>(vkeys, st_kvm, st_kvr, BB, VN, VD);
  bn_apply_relu_kernel<<<(BB*VN*VD + T - 1) / T, T, 0, stream>>>(
      vkeys, st_kvm, st_kvr, kv_g, kv_b, BB*VN*VD, VN, VD);
  pack_kernel<<<(BB*VD*VNP + T - 1) / T, T, 0, stream>>>(vkeys, vquery, Vis, Qt, Kt, Vp);
  rowstats_kernel<<<BB * (VD / 16), 32, 0, stream>>>(Qt, Kt, rowm, rowz);
  vattn_kernel<<<BB * (VD / 64), 128, 0, stream>>>(Qt, Kt, Vp, rowm, rowz, vout);

  // ---------- audio path ----------
  ah_pre_kernel<<<(BB*AN*AN + T - 1) / T, T, 0, stream>>>(Aud, aW1, ab1, ah);
  bn_stats_kernel<<<AN, T, 0, stream>>>(ah, st_a1m, st_a1r, BB, AN, AN);
  bn_apply_relu_kernel<<<(BB*AN*AN + T - 1) / T, T, 0, stream>>>(
      ah, st_a1m, st_a1r, a_g1, a_b1, BB*AN*AN, AN, AN);
  aquery_kernel<<<(BB*AN*AD + T - 1) / T, T, 0, stream>>>(ah, aW2, ab2, aquery);
  akeys_pre_kernel<<<(BB*AN*AD + T - 1) / T, T, 0, stream>>>(Aud, kaW, kab, akeys);
  bn_stats_kernel<<<AN, T, 0, stream>>>(akeys, st_kam, st_kar, BB, AN, AD);
  bn_apply_relu_kernel<<<(BB*AN*AD + T - 1) / T, T, 0, stream>>>(
      akeys, st_kam, st_kar, ka_g, ka_b, BB*AN*AD, AN, AD);
  scores_a_kernel<<<(BB*AD*AD + T - 1) / T, T, 0, stream>>>(akeys, aquery, sca);
  softmax128_kernel<<<BB * AD, 128, 0, stream>>>(sca);
  aout_kernel<<<(BB*AN*AD + T - 1) / T, T, 0, stream>>>(Aud, sca, aout);
}
